// Gemma3nAudioAttention_81449759801607
// MI455X (gfx1250) — matrix operand order repaired and checked by reading, not compiled, not run on hardware
//
#include <hip/hip_runtime.h>
#include <hip/hip_bf16.h>
#include <math.h>

#define HEADS    8
#define HEAD_DIM 192
#define HIDDEN   1536
#define NQKV     4608     // 3 * HEADS * HEAD_DIM
#define CHUNK    12
#define CTX      24
#define PAST     12
#define NFPOS    13       // F = PAST + FUT + 1
#define CAPV     50.0f

// amd_gfx1250_TDM.h ships only with the toolchain whose
// __builtin_amdgcn_tensor_load_to_lds takes 6 args (clang-23 / therock-10.0);
// ROCm 7.2 (clang-22) has the 5-arg form.  Use the header as an arity probe.
#if defined(__has_include)
#if __has_include(<hip/amd_detail/amd_gfx1250_TDM.h>)
#define TDM_ARGS6 1
#else
#define TDM_ARGS6 0
#endif
#else
#define TDM_ARGS6 1
#endif

typedef __attribute__((ext_vector_type(16))) __bf16 v16bf;
typedef __attribute__((ext_vector_type(8)))  __bf16 v8bf;
typedef __attribute__((ext_vector_type(4)))  __bf16 v4bf;
typedef __attribute__((ext_vector_type(8)))  float  v8f;
typedef __attribute__((ext_vector_type(4)))  unsigned int u32x4;
typedef __attribute__((ext_vector_type(8)))  int i32x8;
typedef __attribute__((ext_vector_type(4)))  int i32x4;

union V16 { v16bf v; v8bf h[2]; };   // h[0] = elements 0..7, h[1] = elements 8..15

__device__ inline void f32_hilo(float x, __bf16 &hi, __bf16 &lo) {
    hi = (__bf16)x;
    lo = (__bf16)(x - (float)hi);
}

// ---------------------------------------------------------------------------
// Kernel 0: per-dim q scale + sin_emb = timing @ w_pos  (tiny, VALU fp32)
// ---------------------------------------------------------------------------
__global__ __launch_bounds__(256)
void pos_scale_kernel(const float* __restrict__ w_pos,
                      const float* __restrict__ pds,
                      float* __restrict__ scale,
                      __bf16* __restrict__ sin_emb) {
    int idx = blockIdx.x * 256 + threadIdx.x;
    if (blockIdx.x == 0 && threadIdx.x < HEAD_DIM) {
        float q_scale = rsqrtf((float)HEAD_DIM) / 0.6931471805599453f;
        scale[threadIdx.x] = q_scale * log1pf(expf(pds[threadIdx.x]));
    }
    if (idx < NFPOS * HIDDEN) {
        int f   = idx / HIDDEN;
        int col = idx % HIDDEN;
        float pos = (float)(PAST - f);
        const float neg = -logf(10000.0f) / 767.0f;
        float acc = 0.0f;
        for (int k = 0; k < HIDDEN; ++k) {
            int i = (k < 768) ? k : (k - 768);
            float ang = pos * expf((float)i * neg);
            float tv = (k < 768) ? sinf(ang) : cosf(ang);
            acc = fmaf(tv, w_pos[(size_t)k * HIDDEN + col], acc);
        }
        sin_emb[idx] = (__bf16)acc;               // layout [f][n][h]
    }
}

// ---------------------------------------------------------------------------
// Pre-convert kernels for the TDM path: fp32 -> (hi,lo) bf16.
// X keeps row-major [M][K]; W is transposed to WT[N][K] so A and B tiles are
// both simple strided 2-D tiles for tensor_load_to_lds.
// ---------------------------------------------------------------------------
__global__ __launch_bounds__(256)
void convert_x_kernel(const float* __restrict__ X, __bf16* __restrict__ Xhi,
                      __bf16* __restrict__ Xlo, long long n4) {
    long long i = (long long)blockIdx.x * 256 + threadIdx.x;
    if (i >= n4) return;
    float4 xv = ((const float4*)X)[i];
    v4bf h, l;
    __bf16 hh, ll;
    f32_hilo(xv.x, hh, ll); h[0] = hh; l[0] = ll;
    f32_hilo(xv.y, hh, ll); h[1] = hh; l[1] = ll;
    f32_hilo(xv.z, hh, ll); h[2] = hh; l[2] = ll;
    f32_hilo(xv.w, hh, ll); h[3] = hh; l[3] = ll;
    *(v4bf*)(Xhi + i * 4) = h;
    *(v4bf*)(Xlo + i * 4) = l;
}

__global__ __launch_bounds__(256)
void convert_w_kernel(const float* __restrict__ W, __bf16* __restrict__ WThi,
                      __bf16* __restrict__ WTlo) {
    int idx = blockIdx.x * 256 + threadIdx.x;     // NQKV * (HIDDEN/8) total
    if (idx >= NQKV * (HIDDEN / 8)) return;
    int n  = idx / (HIDDEN / 8);
    int kc = (idx % (HIDDEN / 8)) * 8;
    v8bf h, l;
#pragma unroll
    for (int j = 0; j < 8; ++j) {
        __bf16 hh, ll;
        f32_hilo(W[(size_t)(kc + j) * NQKV + n], hh, ll);
        h[j] = hh; l[j] = ll;
    }
    *(v8bf*)(WThi + (size_t)n * HIDDEN + kc) = h;
    *(v8bf*)(WTlo + (size_t)n * HIDDEN + kc) = l;
}

// ---------------------------------------------------------------------------
// TDM descriptor: load a 128x32 bf16 tile (row stride 1536 elements) into LDS
// with 16B padding per 64B row  ->  80B LDS pitch (= 40 bf16 elements).
// D# per CDNA5 ISA.
// ---------------------------------------------------------------------------
__device__ __forceinline__ void tdm_load_tile(unsigned lds_addr, const void* gp) {
    unsigned long long ga = (unsigned long long)(uintptr_t)gp;
    u32x4 g0 = { 1u,                                  // count=1, user mode
                 lds_addr,                            // LDS byte address
                 (unsigned)(ga & 0xFFFFFFFFu),
                 (unsigned)((ga >> 32) & 0x01FFFFFFull) | 0x80000000u }; // type=2
    i32x8 g1 = { (1 << 16) | (1 << 20) | (3 << 22) | (3 << 25),
                 //  data_size=2B, pad_enable, pad_interval=16DW(64B), pad_amount=4DW(16B)
                 0,                                   // abar=0, tdim0[15:0]=0
                 0x4000,                              // tdim0=1<<30, tdim1[15:0]=0
                 0x4000 | (32 << 16),                 // tdim1=1<<30, tile_dim0=32
                 128,                                 // tile_dim1=128, tile_dim2=0
                 HIDDEN,                              // tensor_dim0_stride = 1536
                 0, 0 };
    i32x4 g2 = { 1, 1, 0, 0 };                        // tensor_dim2=1, tensor_dim3=1
    i32x4 g3 = { 0, 1 << 16, 0, 0 };                  // tensor_dim4=1
#if TDM_ARGS6
    i32x8 g4 = { 0, 0, 0, 0, 0, 0, 0, 0 };
    __builtin_amdgcn_tensor_load_to_lds(g0, g1, g2, g3, g4, 0);
#else
    __builtin_amdgcn_tensor_load_to_lds(g0, g1, g2, g3, 0);
#endif
}

// Shared compute step: 3-term split-bf16 WMMA on one staged 128x128x32 tile.
#define GEMM_COMPUTE(SAH, SAL, SBH, SBL)                                         \
    {                                                                            \
        V16 ah[4], al[4];                                                        \
        _Pragma("unroll")                                                        \
        for (int mt = 0; mt < 4; ++mt) {                                         \
            int r = (wm * 64 + mt * 16 + Ml) * 40;                               \
            ah[mt].h[0] = *(const v8bf*)((SAH) + r + abase);                     \
            ah[mt].h[1] = *(const v8bf*)((SAH) + r + abase + 8);                 \
            al[mt].h[0] = *(const v8bf*)((SAL) + r + abase);                     \
            al[mt].h[1] = *(const v8bf*)((SAL) + r + abase + 8);                 \
        }                                                                        \
        V16 bh[2], bl[2];                                                        \
        _Pragma("unroll")                                                        \
        for (int nt = 0; nt < 2; ++nt) {                                         \
            int r = (wn * 32 + nt * 16 + Ml) * 40;                               \
            bh[nt].h[0] = *(const v8bf*)((SBH) + r + bbase);                     \
            bh[nt].h[1] = *(const v8bf*)((SBH) + r + bbase + 8);                 \
            bl[nt].h[0] = *(const v8bf*)((SBL) + r + bbase);                     \
            bl[nt].h[1] = *(const v8bf*)((SBL) + r + bbase + 8);                 \
        }                                                                        \
        _Pragma("unroll")                                                        \
        for (int mt = 0; mt < 4; ++mt)                                           \
            _Pragma("unroll")                                                    \
            for (int nt = 0; nt < 2; ++nt) {                                     \
                acc[mt][nt] = __builtin_amdgcn_wmma_f32_16x16x32_bf16(           \
                    false, ah[mt].v, false, bh[nt].v, (short)0, acc[mt][nt], false, false); \
                acc[mt][nt] = __builtin_amdgcn_wmma_f32_16x16x32_bf16(           \
                    false, ah[mt].v, false, bl[nt].v, (short)0, acc[mt][nt], false, false); \
                acc[mt][nt] = __builtin_amdgcn_wmma_f32_16x16x32_bf16(           \
                    false, al[mt].v, false, bh[nt].v, (short)0, acc[mt][nt], false, false); \
            }                                                                    \
    }

#define GEMM_EPILOGUE()                                                          \
    {                                                                            \
        const int Mh = (lane >= 16) ? 8 : 0;                                     \
        _Pragma("unroll")                                                        \
        for (int nt = 0; nt < 2; ++nt) {                                         \
            int Ng  = n0 + wn * 32 + nt * 16 + Ml;                               \
            int sec = Ng / HIDDEN;                                               \
            int c2  = Ng % HIDDEN;                                               \
            float s = (sec == 0) ? scale[c2 % HEAD_DIM] : 1.0f;                  \
            __bf16* dst = (sec == 0) ? Q : ((sec == 1) ? K : V);                 \
            _Pragma("unroll")                                                    \
            for (int mt = 0; mt < 4; ++mt) {                                     \
                int Mg0 = m0 + wm * 64 + mt * 16 + Mh;                           \
                _Pragma("unroll")                                                \
                for (int r = 0; r < 8; ++r)                                      \
                    dst[(size_t)(Mg0 + r) * HIDDEN + c2] =                       \
                        (__bf16)(acc[mt][nt][r] * s);                            \
            }                                                                    \
        }                                                                        \
    }

// ---------------------------------------------------------------------------
// Kernel 1a: QKV GEMM, TDM-staged + double-buffered.
// Wave 0 issues 4 tensor_load_to_lds per K-step; compute waves only do
// ds_load_b128 + WMMA.  Sync: s_wait_tensorcnt + workgroup barrier.
// ---------------------------------------------------------------------------
__global__ __launch_bounds__(256)
void qkv_gemm_tdm_kernel(const __bf16* __restrict__ Xhi, const __bf16* __restrict__ Xlo,
                         const __bf16* __restrict__ WThi, const __bf16* __restrict__ WTlo,
                         const float* __restrict__ scale,
                         __bf16* __restrict__ Q, __bf16* __restrict__ K,
                         __bf16* __restrict__ V) {
    __shared__ __bf16 smem[2][4][128 * 40];   // [buf][Ahi,Alo,Bhi,Blo]

    const int tid  = threadIdx.x;
    const int lane = tid & 31;
    const int w    = tid >> 5;
    const int wm   = w >> 2;
    const int wn   = w & 3;
    const int m0   = blockIdx.y * 128;
    const int n0   = blockIdx.x * 128;

    v8f acc[4][2];
#pragma unroll
    for (int mt = 0; mt < 4; ++mt)
#pragma unroll
        for (int nt = 0; nt < 2; ++nt)
#pragma unroll
            for (int e = 0; e < 8; ++e) acc[mt][nt][e] = 0.0f;

    const int Ml    = lane & 15;
    const int abase = (lane < 16) ? 0 : 16;
    const int bbase = (lane < 16) ? 0 : 16;

    const __bf16* gsrc[4] = { Xhi  + (size_t)m0 * HIDDEN,
                              Xlo  + (size_t)m0 * HIDDEN,
                              WThi + (size_t)n0 * HIDDEN,
                              WTlo + (size_t)n0 * HIDDEN };

    if (w == 0) {
#pragma unroll
        for (int a = 0; a < 4; ++a)
            tdm_load_tile((unsigned)(uintptr_t)(&smem[0][a][0]), gsrc[a]);
    }

    const int NK = HIDDEN / 32;
    for (int t = 0; t < NK; ++t) {
        if (w == 0) __builtin_amdgcn_s_wait_tensorcnt(0);
        __syncthreads();                       // buf[t&1] visible to all waves
        if (w == 0 && t + 1 < NK) {
#pragma unroll
            for (int a = 0; a < 4; ++a)
                tdm_load_tile((unsigned)(uintptr_t)(&smem[(t + 1) & 1][a][0]),
                              gsrc[a] + (size_t)(t + 1) * 32);
        }
        const __bf16* sAh = smem[t & 1][0];
        const __bf16* sAl = smem[t & 1][1];
        const __bf16* sBh = smem[t & 1][2];
        const __bf16* sBl = smem[t & 1][3];
        GEMM_COMPUTE(sAh, sAl, sBh, sBl);
    }
    GEMM_EPILOGUE();
}

// ---------------------------------------------------------------------------
// Kernel 1b (fallback): QKV GEMM with in-loop fp32->bf16 hi/lo staging.
// ---------------------------------------------------------------------------
__global__ __launch_bounds__(256)
void qkv_gemm_kernel(const float* __restrict__ X, const float* __restrict__ W,
                     const float* __restrict__ scale,
                     __bf16* __restrict__ Q, __bf16* __restrict__ K,
                     __bf16* __restrict__ V) {
    __shared__ __bf16 sAh[128 * 40];
    __shared__ __bf16 sAl[128 * 40];
    __shared__ __bf16 sBh[128 * 40];
    __shared__ __bf16 sBl[128 * 40];

    const int tid  = threadIdx.x;
    const int lane = tid & 31;
    const int w    = tid >> 5;
    const int wm   = w >> 2;
    const int wn   = w & 3;
    const int m0   = blockIdx.y * 128;
    const int n0   = blockIdx.x * 128;

    v8f acc[4][2];
#pragma unroll
    for (int mt = 0; mt < 4; ++mt)
#pragma unroll
        for (int nt = 0; nt < 2; ++nt)
#pragma unroll
            for (int e = 0; e < 8; ++e) acc[mt][nt][e] = 0.0f;

    const int Ml    = lane & 15;
    const int abase = (lane < 16) ? 0 : 16;
    const int bbase = (lane < 16) ? 0 : 16;

    for (int k0 = 0; k0 < HIDDEN; k0 += 32) {
#pragma unroll
        for (int i = 0; i < 4; ++i) {
            int row = (tid >> 3) + i * 32;
            int kc  = (tid & 7) * 4;
            float4 xv = *(const float4*)(X + (size_t)(m0 + row) * HIDDEN + k0 + kc);
            if (k0 + 32 < HIDDEN)
                __builtin_prefetch(X + (size_t)(m0 + row) * HIDDEN + k0 + 32 + kc, 0, 0);
            __bf16 h0, l0;
            f32_hilo(xv.x, h0, l0); sAh[row*40+kc+0]=h0; sAl[row*40+kc+0]=l0;
            f32_hilo(xv.y, h0, l0); sAh[row*40+kc+1]=h0; sAl[row*40+kc+1]=l0;
            f32_hilo(xv.z, h0, l0); sAh[row*40+kc+2]=h0; sAl[row*40+kc+2]=l0;
            f32_hilo(xv.w, h0, l0); sAh[row*40+kc+3]=h0; sAl[row*40+kc+3]=l0;
        }
        {
            int n  = tid & 127;
            int kg = (tid >> 7) * 16;
#pragma unroll
            for (int j = 0; j < 16; ++j) {
                float wv = W[(size_t)(k0 + kg + j) * NQKV + n0 + n];
                __bf16 h0, l0;
                f32_hilo(wv, h0, l0);
                sBh[n*40 + kg + j] = h0;
                sBl[n*40 + kg + j] = l0;
            }
        }
        __syncthreads();
        GEMM_COMPUTE(sAh, sAl, sBh, sBl);
        __syncthreads();
    }
    GEMM_EPILOGUE();
}

// ---------------------------------------------------------------------------
// Kernel 2: fused local attention.  One wave per (b, u, head).
// ---------------------------------------------------------------------------
__global__ __launch_bounds__(32)
void attn_kernel(const __bf16* __restrict__ Q, const __bf16* __restrict__ K,
                 const __bf16* __restrict__ V, const __bf16* __restrict__ sin_emb,
                 const unsigned char* __restrict__ mask,
                 float* __restrict__ out, int B, int T) {
    __shared__ __bf16 vbT[HEAD_DIM * 32];
    __shared__ float  lgt[16 * 32];
    __shared__ float  bdl[16 * 16];
    __shared__ __bf16 probs[16 * 32];

    const int lane = threadIdx.x;
    int id = blockIdx.x;
    const int n = id % HEADS;  id /= HEADS;
    const int U = T / CHUNK;
    const int u = id % U;      id /= U;
    const int b = id;
    const int t0 = u * CHUNK - PAST;

    const int Ml    = lane & 15;
    const int Mh    = (lane >= 16) ? 8 : 0;
    const int abase = (lane < 16) ? 0 : 16;
    const int bbase = (lane < 16) ? 0 : 16;

    v8bf zv;
#pragma unroll
    for (int e = 0; e < 8; ++e) zv[e] = (__bf16)0.0f;

    for (int i = lane; i < HEAD_DIM * 32 / 8; i += 32)
        *(v8bf*)(vbT + i * 8) = zv;
    __syncthreads();
    for (int i = lane; i < CTX * (HEAD_DIM / 8); i += 32) {
        int c  = i / (HEAD_DIM / 8);
        int h8 = (i % (HEAD_DIM / 8)) * 8;
        int t  = t0 + c;
        v8bf val = zv;
        if (t >= 0 && t < T)
            val = *(const v8bf*)(V + ((size_t)(b * T + t) * HEADS + n) * HEAD_DIM + h8);
#pragma unroll
        for (int e = 0; e < 8; ++e) vbT[(h8 + e) * 32 + c] = val[e];
    }

    v8f acc0, acc1, accbd;
#pragma unroll
    for (int e = 0; e < 8; ++e) { acc0[e] = 0.f; acc1[e] = 0.f; accbd[e] = 0.f; }

    const size_t qrow0 = ((size_t)(b * T + u * CHUNK) * HEADS + n) * HEAD_DIM;
#pragma unroll
    for (int kt = 0; kt < 6; ++kt) {
        V16 aq;
        if (Ml < CHUNK) {
            const __bf16* p = Q + qrow0 + (size_t)Ml * HEADS * HEAD_DIM + kt * 32 + abase;
            aq.h[0] = *(const v8bf*)(p);
            aq.h[1] = *(const v8bf*)(p + 8);
        } else { aq.h[0] = zv; aq.h[1] = zv; }

#pragma unroll
        for (int nt = 0; nt < 2; ++nt) {
            V16 bk;
            int c = nt * 16 + Ml;
            int t = t0 + c;
            if (c < CTX && t >= 0 && t < T) {
                const __bf16* p = K + ((size_t)(b * T + t) * HEADS + n) * HEAD_DIM + kt * 32 + bbase;
                bk.h[0] = *(const v8bf*)(p);
                bk.h[1] = *(const v8bf*)(p + 8);
            } else { bk.h[0] = zv; bk.h[1] = zv; }
            if (nt == 0)
                acc0 = __builtin_amdgcn_wmma_f32_16x16x32_bf16(
                    false, aq.v, false, bk.v, (short)0, acc0, false, false);
            else
                acc1 = __builtin_amdgcn_wmma_f32_16x16x32_bf16(
                    false, aq.v, false, bk.v, (short)0, acc1, false, false);
        }
        V16 bs;
        if (Ml < NFPOS) {
            const __bf16* p = sin_emb + ((size_t)Ml * HEADS + n) * HEAD_DIM + kt * 32 + bbase;
            bs.h[0] = *(const v8bf*)(p);
            bs.h[1] = *(const v8bf*)(p + 8);
        } else { bs.h[0] = zv; bs.h[1] = zv; }
        accbd = __builtin_amdgcn_wmma_f32_16x16x32_bf16(
            false, aq.v, false, bs.v, (short)0, accbd, false, false);
    }

#pragma unroll
    for (int r = 0; r < 8; ++r) {
        lgt[(Mh + r) * 32 + Ml]      = acc0[r];
        lgt[(Mh + r) * 32 + 16 + Ml] = acc1[r];
        bdl[(Mh + r) * 16 + Ml]      = accbd[r];
    }
    __syncthreads();

    if (lane < 16) {
        int i = lane;
        if (i < CHUNK) {
            float l[CTX];
            float mx = -3.4e38f;
            for (int j = 0; j < CTX; ++j) {
                int t = t0 + j;
                bool causal = (j >= i) && (j <= i + PAST);
                bool valid  = causal && (t >= 0) && (t < T) && (mask[b * T + t] == 0);
                float lv = lgt[i * 32 + j];
                if (causal) lv += bdl[i * 16 + (j - i)];
                lv = CAPV * tanhf(lv * (1.0f / CAPV));
                lv = valid ? lv : -3.4e38f;
                l[j] = lv;
                mx = fmaxf(mx, lv);
            }
            float sum = 0.0f;
            for (int j = 0; j < CTX; ++j) {
                float p = (l[j] <= -3.3e38f) ? 0.0f : __expf(l[j] - mx);
                l[j] = p;
                sum += p;
            }
            float inv = (sum > 0.0f) ? 1.0f / sum : 0.0f;
            for (int j = 0; j < 32; ++j)
                probs[i * 32 + j] = (__bf16)((j < CTX) ? l[j] * inv : 0.0f);
        } else {
            for (int j = 0; j < 32; ++j) probs[i * 32 + j] = (__bf16)0.0f;
        }
    }
    __syncthreads();

    V16 ap;
    ap.h[0] = *(const v8bf*)(probs + Ml * 32 + abase);
    ap.h[1] = *(const v8bf*)(probs + Ml * 32 + abase + 8);

#pragma unroll
    for (int nt = 0; nt < HEAD_DIM / 16; ++nt) {
        V16 bv;
        const __bf16* p = vbT + (nt * 16 + Ml) * 32 + bbase;
        bv.h[0] = *(const v8bf*)(p);
        bv.h[1] = *(const v8bf*)(p + 8);
        v8f o;
#pragma unroll
        for (int e = 0; e < 8; ++e) o[e] = 0.f;
        o = __builtin_amdgcn_wmma_f32_16x16x32_bf16(
            false, ap.v, false, bv.v, (short)0, o, false, false);
#pragma unroll
        for (int r = 0; r < 8; ++r) {
            int M = Mh + r;
            if (M < CHUNK)
                out[((size_t)(b * T + u * CHUNK + M) * HEADS + n) * HEAD_DIM + nt * 16 + Ml] = o[r];
        }
    }
}

// ---------------------------------------------------------------------------
// Host launch. ws: scale(1KB) | sin_emb(40KB) | Q | K | V [| Xhi | Xlo | WThi | WTlo]
// ---------------------------------------------------------------------------
extern "C" void kernel_launch(void* const* d_in, const int* in_sizes, int n_in,
                              void* d_out, int out_size, void* d_ws, size_t ws_size,
                              hipStream_t stream) {
    const float*         x     = (const float*)d_in[0];
    const unsigned char* mask  = (const unsigned char*)d_in[1];
    const float*         w_qkv = (const float*)d_in[2];
    const float*         w_pos = (const float*)d_in[3];
    const float*         pds   = (const float*)d_in[4];

    size_t BT = (size_t)in_sizes[1];
    int T = 6144;
    int B = (int)(BT / T);
    if ((size_t)B * T != BT) { B = 1; T = (int)BT; }

    char* ws = (char*)d_ws;
    float*  scale   = (float*)ws;
    __bf16* sin_emb = (__bf16*)(ws + 1024);
    __bf16* Qb      = (__bf16*)(ws + 1024 + 40960);
    __bf16* Kb      = Qb + BT * HIDDEN;
    __bf16* Vb      = Kb + BT * HIDDEN;
    __bf16* Xhi     = Vb + BT * HIDDEN;
    __bf16* Xlo     = Xhi + BT * HIDDEN;
    __bf16* WThi    = Xlo + BT * HIDDEN;
    __bf16* WTlo    = WThi + (size_t)HIDDEN * NQKV;

    size_t need_fb  = 1024 + 40960 + 3 * BT * HIDDEN * 2;
    size_t need_tdm = need_fb + 2 * BT * HIDDEN * 2 + 2 * (size_t)HIDDEN * NQKV * 2;

    pos_scale_kernel<<<(NFPOS * HIDDEN + 255) / 256, 256, 0, stream>>>(
        w_pos, pds, scale, sin_emb);

    dim3 ggrid(NQKV / 128, (unsigned)(BT / 128));
    if (ws_size >= need_tdm && (BT % 128) == 0) {
        long long n4 = (long long)(BT * HIDDEN / 4);
        convert_x_kernel<<<(unsigned)((n4 + 255) / 256), 256, 0, stream>>>(x, Xhi, Xlo, n4);
        convert_w_kernel<<<(NQKV * (HIDDEN / 8) + 255) / 256, 256, 0, stream>>>(w_qkv, WThi, WTlo);
        qkv_gemm_tdm_kernel<<<ggrid, 256, 0, stream>>>(Xhi, Xlo, WThi, WTlo, scale, Qb, Kb, Vb);
    } else {
        qkv_gemm_kernel<<<ggrid, 256, 0, stream>>>(x, w_qkv, scale, Qb, Kb, Vb);
    }

    attn_kernel<<<(unsigned)(BT / CHUNK) * HEADS, 32, 0, stream>>>(
        Qb, Kb, Vb, sin_emb, mask, (float*)d_out, B, T);
}
